// LJMM_82471962018254
// MI455X (gfx1250) — matrix operand order repaired and checked by reading, not compiled, not run on hardware
//
#include <hip/hip_runtime.h>
#include <stdint.h>
#include <stddef.h>

typedef __bf16 bf16_t;
typedef __attribute__((ext_vector_type(16))) __bf16 v16bf;
typedef __attribute__((ext_vector_type(8)))  __bf16 v8bf;
typedef __attribute__((ext_vector_type(4)))  __bf16 v4bf;
typedef __attribute__((ext_vector_type(8)))  float  v8f;

#define BLK_M 64
#define BLK_N 64
#define BLK_K 32
#define LDA   40   // BLK_K + 8 bf16 pad -> 80B row stride (16B aligned)

__device__ __forceinline__ bf16_t f2bf(float x) { return (bf16_t)x; }

__device__ __forceinline__ v8bf zero_v8bf() {
  v8bf z;
#pragma unroll
  for (int i = 0; i < 8; ++i) z[i] = (bf16_t)0.0f;
  return z;
}

// ---------------------------------------------------------------------------
// WMMA tile core: 4 waves, each a 32x32 sub-tile (2x2 of 16x16x32 bf16 WMMA
// fragments) of the 64x64 block. A staged row-major [64][LDA]; B staged
// TRANSPOSED [n][LDA] so all fragment loads are contiguous ds_load_b128,
// matching the CDNA5 16-bit A(16x32) / B(32x16) VGPR layouts.
// ---------------------------------------------------------------------------
__device__ __forceinline__ void wmma_tile(const bf16_t* Al, const bf16_t* Bl,
                                          v8f (&acc)[2][2]) {
  const int lane = threadIdx.x & 31;
  const int wave = threadIdx.x >> 5;
  const int wm = (wave >> 1) * 32;
  const int wn = (wave & 1) * 32;
  const int r  = lane & 15;
  const int hi = lane >> 4;
  v16bf afrag[2], bfrag[2];
#pragma unroll
  for (int fa = 0; fa < 2; ++fa) {
    const bf16_t* base = Al + (wm + fa * 16 + r) * LDA;
    v8bf lo = *(const v8bf*)(base + hi * 8);        // K = hi*8 .. +7
    v8bf hh = *(const v8bf*)(base + 16 + hi * 8);   // K = 16+hi*8 .. +7
#pragma unroll
    for (int i = 0; i < 8; ++i) { afrag[fa][i] = lo[i]; afrag[fa][8 + i] = hh[i]; }
  }
#pragma unroll
  for (int fb = 0; fb < 2; ++fb) {
    const bf16_t* base = Bl + (wn + fb * 16 + r) * LDA + hi * 8;
    v8bf lo = *(const v8bf*)(base);                 // K = hi*8 .. +7
    v8bf hh = *(const v8bf*)(base + 16);            // K = 16+hi*8 .. +7
#pragma unroll
    for (int i = 0; i < 8; ++i) { bfrag[fb][i] = lo[i]; bfrag[fb][8 + i] = hh[i]; }
  }
#pragma unroll
  for (int fa = 0; fa < 2; ++fa)
#pragma unroll
    for (int fb = 0; fb < 2; ++fb)
      acc[fa][fb] = __builtin_amdgcn_wmma_f32_16x16x32_bf16(
          false, afrag[fa], false, bfrag[fb], (short)0, acc[fa][fb], false, false);
}

__device__ __forceinline__ void zero_acc(v8f (&acc)[2][2]) {
#pragma unroll
  for (int a = 0; a < 2; ++a)
#pragma unroll
    for (int b = 0; b < 2; ++b)
#pragma unroll
      for (int j = 0; j < 8; ++j) acc[a][b][j] = 0.0f;
}

// --------------------- vectorized tile fetch / stage helpers ----------------
// B operand (bf16, row n contiguous over k, row stride sK elements):
// thread -> (kk8 = (tid&3)*8, rows nn and nn+32): two 16B loads / stores.
struct BReg { v8bf b0, b1; };
__device__ __forceinline__ BReg loadB(const bf16_t* __restrict__ Bsrc, int sK,
                                      int n0, int kb, int tid) {
  const int kk8 = (tid & 3) * 8, nn = tid >> 2;
  BReg r;
  r.b0 = *(const v8bf*)(Bsrc + (size_t)(n0 + nn) * sK + kb + kk8);
  r.b1 = *(const v8bf*)(Bsrc + (size_t)(n0 + nn + 32) * sK + kb + kk8);
  return r;
}
__device__ __forceinline__ void storeB(bf16_t* Blds, const BReg& r, int tid) {
  const int kk8 = (tid & 3) * 8, nn = tid >> 2;
  *(v8bf*)(Blds + nn * LDA + kk8) = r.b0;
  *(v8bf*)(Blds + (nn + 32) * LDA + kk8) = r.b1;
}

// A operand from a row-major bf16 matrix (Hbf/Xbf style), optional row shift
// with zero padding (for the conv im2col).
struct AReg { v8bf a0, a1; };
__device__ __forceinline__ void storeA(bf16_t* Alds, const AReg& r, int tid) {
  const int kk8 = (tid & 3) * 8, rr = tid >> 2;
  *(v8bf*)(Alds + rr * LDA + kk8) = r.a0;
  *(v8bf*)(Alds + (rr + 32) * LDA + kk8) = r.a1;
}

// A operand from f32 source (combine kernel): float4 -> packed v4bf.
struct AF32Reg { float4 a[4]; };
__device__ __forceinline__ AF32Reg loadAf32(const float* __restrict__ src,
                                            int kb, int tid) {
  const int kk4 = (tid & 7) * 4, rr = tid >> 3;  // rows rr, rr+16, rr+32, rr+48
  AF32Reg r;
#pragma unroll
  for (int q = 0; q < 4; ++q)
    r.a[q] = *(const float4*)(src + (size_t)(rr + q * 16) * 512 + kb + kk4);
  return r;
}
__device__ __forceinline__ void storeAf32(bf16_t* Alds, const AF32Reg& r, int tid) {
  const int kk4 = (tid & 7) * 4, rr = tid >> 3;
#pragma unroll
  for (int q = 0; q < 4; ++q) {
    v4bf p;
    p[0] = f2bf(r.a[q].x); p[1] = f2bf(r.a[q].y);
    p[2] = f2bf(r.a[q].z); p[3] = f2bf(r.a[q].w);
    *(v4bf*)(Alds + (rr + q * 16) * LDA + kk4) = p;
  }
}

// ---------------------------------------------------------------------------
// Kernel 1: fused front GEMM (K=640 folded weight).
// shared[b,t,n] = sum_k Afront[b,t,k]*WtF[n,k] + b_total[n]; epilogue fuses
// ReLU + BN and stores bf16 H. A gathers audio/video with on-the-fly
// transpose; the source select is uniform per 32-wide K block.
// ---------------------------------------------------------------------------
struct AFrontReg { float4 a[4]; };

__global__ void __launch_bounds__(128) front_gemm(
    const float* __restrict__ audio, const float* __restrict__ video,
    const bf16_t* __restrict__ WtF, const float* __restrict__ b_total,
    const float* __restrict__ gamma, const float* __restrict__ beta,
    bf16_t* __restrict__ Hbf) {
  __shared__ __align__(16) bf16_t Alds[BLK_M * LDA];
  __shared__ __align__(16) bf16_t Blds[BLK_N * LDA];
  const int row0 = blockIdx.x * BLK_M;       // b*1024 + t0
  const int bb   = row0 >> 10;
  const int t0   = row0 & 1023;
  const int n0   = blockIdx.y * BLK_N;
  const int tid  = threadIdx.x;
  const int r4 = (tid & 15) * 4, kkc = tid >> 4;   // A-fill mapping
  v8f acc[2][2]; zero_acc(acc);

  const float* abase = audio + (size_t)bb * 128 * 1024 + t0 + r4;
  const float* vbase = video + (size_t)bb * 512 * 1024 + t0 + r4;

  AFrontReg ar; BReg br;
  // prologue: fetch K-block 0
#pragma unroll
  for (int q = 0; q < 4; ++q)
    ar.a[q] = *(const float4*)(abase + (size_t)(kkc + q * 8) * 1024);
  br = loadB(WtF, 640, n0, 0, tid);

  for (int kb = 0; kb < 640; kb += BLK_K) {
    // stage current regs -> LDS
#pragma unroll
    for (int q = 0; q < 4; ++q) {
      const int c = kkc + q * 8;
      Alds[(r4 + 0) * LDA + c] = f2bf(ar.a[q].x);
      Alds[(r4 + 1) * LDA + c] = f2bf(ar.a[q].y);
      Alds[(r4 + 2) * LDA + c] = f2bf(ar.a[q].z);
      Alds[(r4 + 3) * LDA + c] = f2bf(ar.a[q].w);
    }
    storeB(Blds, br, tid);
    __syncthreads();
    // prefetch next K-block while WMMAs run
    const int nkb = kb + BLK_K;
    if (nkb < 640) {
      const float* src = (nkb < 128) ? abase + (size_t)nkb * 1024
                                     : vbase + (size_t)(nkb - 128) * 1024;
#pragma unroll
      for (int q = 0; q < 4; ++q)
        ar.a[q] = *(const float4*)(src + (size_t)(kkc + q * 8) * 1024);
      br = loadB(WtF, 640, n0, nkb, tid);
    }
    wmma_tile(Alds, Blds, acc);
    __syncthreads();
  }

  const int lane = tid & 31, wave = tid >> 5;
  const int wm = (wave >> 1) * 32, wn = (wave & 1) * 32;
  const int hi = lane >> 4, nl = lane & 15;
  const float rs = rsqrtf(1.0f + 1e-5f);
#pragma unroll
  for (int fa = 0; fa < 2; ++fa)
#pragma unroll
    for (int fb = 0; fb < 2; ++fb) {
      const int n = n0 + wn + fb * 16 + nl;
      const float g = gamma[n] * rs, bt = beta[n], bo = b_total[n];
#pragma unroll
      for (int j = 0; j < 8; ++j) {
        const int m = row0 + wm + fa * 16 + hi * 8 + j;
        const float s = acc[fa][fb][j] + bo;
        Hbf[(size_t)m * 512 + n] = f2bf(fmaxf(s, 0.0f) * g + bt);
      }
    }
}

// ---------------------------------------------------------------------------
// Kernel 2: conv1d(k=3, SAME) as im2col GEMM, K = 3*512 = 1536.
// A[r, kc*512+i] = H[b, t0+r+kc-1, i], zero-padded at sequence edges.
// ---------------------------------------------------------------------------
__global__ void __launch_bounds__(128) conv_gemm(
    const bf16_t* __restrict__ Hbf, const bf16_t* __restrict__ WrT,
    const float* __restrict__ b_res, bf16_t* __restrict__ Xbf) {
  __shared__ __align__(16) bf16_t Alds[BLK_M * LDA];
  __shared__ __align__(16) bf16_t Blds[BLK_N * LDA];
  const int row0 = blockIdx.x * BLK_M;
  const int bb   = row0 >> 10;
  const int t0   = row0 & 1023;
  const int n0   = blockIdx.y * BLK_N;
  const int tid  = threadIdx.x;
  const int kk8 = (tid & 3) * 8, rr = tid >> 2;
  const bf16_t* Hb = Hbf + (size_t)bb * 1024 * 512;
  v8f acc[2][2]; zero_acc(acc);

  AReg ar; BReg br;
  auto loadA = [&](int kb) {
    AReg r;
    const int kc = kb >> 9;                 // conv tap, uniform per K-block
    const int i0 = (kb & 511) + kk8;
    const int st0 = t0 + rr + kc - 1;
    const int st1 = st0 + 32;
    r.a0 = (st0 >= 0 && st0 < 1024) ? *(const v8bf*)(Hb + (size_t)st0 * 512 + i0)
                                    : zero_v8bf();
    r.a1 = (st1 >= 0 && st1 < 1024) ? *(const v8bf*)(Hb + (size_t)st1 * 512 + i0)
                                    : zero_v8bf();
    return r;
  };
  ar = loadA(0);
  br = loadB(WrT, 1536, n0, 0, tid);

  for (int kb = 0; kb < 1536; kb += BLK_K) {
    storeA(Alds, ar, tid);
    storeB(Blds, br, tid);
    __syncthreads();
    const int nkb = kb + BLK_K;
    if (nkb < 1536) { ar = loadA(nkb); br = loadB(WrT, 1536, n0, nkb, tid); }
    wmma_tile(Alds, Blds, acc);
    __syncthreads();
  }

  const int lane = tid & 31, wave = tid >> 5;
  const int wm = (wave >> 1) * 32, wn = (wave & 1) * 32;
  const int hi = lane >> 4, nl = lane & 15;
#pragma unroll
  for (int fa = 0; fa < 2; ++fa)
#pragma unroll
    for (int fb = 0; fb < 2; ++fb) {
      const int n = n0 + wn + fb * 16 + nl;
      const float brv = b_res[n];
#pragma unroll
      for (int j = 0; j < 8; ++j) {
        const int m = row0 + wm + fa * 16 + hi * 8 + j;
        Xbf[(size_t)m * 512 + n] = f2bf(acc[fa][fb][j] + brv);
      }
    }
}

// ---------------------------------------------------------------------------
// Kernel 3: U = X @ B_in, stored reordered as U[t][b][n] (f32) for the fold.
// ---------------------------------------------------------------------------
__global__ void __launch_bounds__(128) u_gemm(
    const bf16_t* __restrict__ Xbf, const bf16_t* __restrict__ BinT,
    float* __restrict__ U) {
  __shared__ __align__(16) bf16_t Alds[BLK_M * LDA];
  __shared__ __align__(16) bf16_t Blds[BLK_N * LDA];
  const int row0 = blockIdx.x * BLK_M;
  const int n0   = blockIdx.y * BLK_N;
  const int tid  = threadIdx.x;
  const int kk8 = (tid & 3) * 8, rr = tid >> 2;
  const bf16_t* Xb = Xbf + (size_t)row0 * 512 + kk8;
  v8f acc[2][2]; zero_acc(acc);

  AReg ar; BReg br;
  ar.a0 = *(const v8bf*)(Xb + (size_t)rr * 512);
  ar.a1 = *(const v8bf*)(Xb + (size_t)(rr + 32) * 512);
  br = loadB(BinT, 512, n0, 0, tid);

  for (int kb = 0; kb < 512; kb += BLK_K) {
    storeA(Alds, ar, tid);
    storeB(Blds, br, tid);
    __syncthreads();
    const int nkb = kb + BLK_K;
    if (nkb < 512) {
      ar.a0 = *(const v8bf*)(Xb + (size_t)rr * 512 + nkb);
      ar.a1 = *(const v8bf*)(Xb + (size_t)(rr + 32) * 512 + nkb);
      br = loadB(BinT, 512, n0, nkb, tid);
    }
    wmma_tile(Alds, Blds, acc);
    __syncthreads();
  }

  const int lane = tid & 31, wave = tid >> 5;
  const int wm = (wave >> 1) * 32, wn = (wave & 1) * 32;
  const int hi = lane >> 4, nl = lane & 15;
#pragma unroll
  for (int fa = 0; fa < 2; ++fa)
#pragma unroll
    for (int fb = 0; fb < 2; ++fb) {
      const int n = n0 + wn + fb * 16 + nl;
#pragma unroll
      for (int j = 0; j < 8; ++j) {
        const int m = row0 + wm + fa * 16 + hi * 8 + j;
        const int t = m & 1023, b2 = m >> 10;
        U[((size_t)t * 64 + b2) * 512 + n] = acc[fa][fb][j];
      }
    }
}

// ---------------------------------------------------------------------------
// Kernel 4: tree-reduction combine for the recurrence fold.
// Vdst[s] = Vsrc[2s] @ M + Vsrc[2s+1]  (V segments [64,512] f32, M = A^(2^lvl)
// supplied transposed bf16). grid = (nseg, 8).
// ---------------------------------------------------------------------------
__global__ void __launch_bounds__(128) combine_gemm(
    const float* __restrict__ Vsrc, const bf16_t* __restrict__ MT,
    float* __restrict__ Vdst) {
  __shared__ __align__(16) bf16_t Alds[BLK_M * LDA];
  __shared__ __align__(16) bf16_t Blds[BLK_N * LDA];
  const int s  = blockIdx.x;
  const int n0 = blockIdx.y * BLK_N;
  const int tid = threadIdx.x;
  const float* Aev = Vsrc + (size_t)(2 * s) * 64 * 512;
  const float* Aod = Vsrc + (size_t)(2 * s + 1) * 64 * 512;
  v8f acc[2][2]; zero_acc(acc);

  AF32Reg ar = loadAf32(Aev, 0, tid);
  BReg br = loadB(MT, 512, n0, 0, tid);

  for (int kb = 0; kb < 512; kb += BLK_K) {
    storeAf32(Alds, ar, tid);
    storeB(Blds, br, tid);
    __syncthreads();
    const int nkb = kb + BLK_K;
    if (nkb < 512) { ar = loadAf32(Aev, nkb, tid); br = loadB(MT, 512, n0, nkb, tid); }
    wmma_tile(Alds, Blds, acc);
    __syncthreads();
  }

  const int lane = tid & 31, wave = tid >> 5;
  const int wm = (wave >> 1) * 32, wn = (wave & 1) * 32;
  const int hi = lane >> 4, nl = lane & 15;
#pragma unroll
  for (int fa = 0; fa < 2; ++fa)
#pragma unroll
    for (int fb = 0; fb < 2; ++fb) {
      const int n = n0 + wn + fb * 16 + nl;
#pragma unroll
      for (int j = 0; j < 8; ++j) {
        const int m = wm + fa * 16 + hi * 8 + j;
        Vdst[(size_t)s * 64 * 512 + (size_t)m * 512 + n] =
            acc[fa][fb][j] + Aod[(size_t)m * 512 + n];
      }
    }
}

// ---------------------------------------------------------------------------
// Small helper kernels (weight folding, packing, bias, final projection).
// ---------------------------------------------------------------------------
__global__ void gemm_f32(const float* __restrict__ A, const float* __restrict__ B,
                         float* __restrict__ C, int M, int N, int K,
                         int sAm, int sAk, int sBk, int sBn) {
  const int idx = blockIdx.x * 256 + threadIdx.x;
  if (idx >= M * N) return;
  const int m = idx / N, n = idx % N;
  float acc = 0.0f;
  for (int k = 0; k < K; ++k)
    acc += A[(size_t)m * sAm + (size_t)k * sAk] * B[(size_t)k * sBk + (size_t)n * sBn];
  C[(size_t)m * N + n] = acc;
}

// dst[c*rows + r] = bf16(src[r*cols + c])
__global__ void packT_bf16(const float* __restrict__ src, bf16_t* __restrict__ dst,
                           int rows, int cols) {
  const int idx = blockIdx.x * 256 + threadIdx.x;
  if (idx >= rows * cols) return;
  const int c = idx / rows, r = idx % rows;
  dst[idx] = f2bf(src[(size_t)r * cols + c]);
}

__global__ void pack_frontT(const float* __restrict__ Ua, const float* __restrict__ Uv,
                            bf16_t* __restrict__ WtF) {
  const int idx = blockIdx.x * 256 + threadIdx.x;
  if (idx >= 512 * 640) return;
  const int n = idx / 640, k = idx % 640;
  const float v = (k < 128) ? Ua[(size_t)k * 512 + n] : Uv[(size_t)(k - 128) * 512 + n];
  WtF[idx] = f2bf(v);
}

__global__ void pack_convT(const float* __restrict__ w_res, bf16_t* __restrict__ WrT) {
  const int idx = blockIdx.x * 256 + threadIdx.x;
  if (idx >= 512 * 1536) return;
  const int n = idx / 1536, k = idx % 1536;
  const int kc = k >> 9, i = k & 511;
  WrT[idx] = f2bf(w_res[((size_t)n * 512 + i) * 3 + kc]);
}

// out[l] = sum_h bvec[h]*W[h*L + l] + badd[l]
__global__ void bias_mm(const float* __restrict__ bvec, const float* __restrict__ W,
                        const float* __restrict__ badd, float* __restrict__ out,
                        int H, int L) {
  const int l = blockIdx.x * 256 + threadIdx.x;
  if (l >= L) return;
  float acc = badd[l];
  for (int h = 0; h < H; ++h) acc += bvec[h] * W[(size_t)h * L + l];
  out[l] = acc;
}

__global__ void bias_combine(const float* __restrict__ tba, const float* __restrict__ tbv,
                             const float* __restrict__ Ws, const float* __restrict__ bsh,
                             float* __restrict__ btot) {
  const int l = blockIdx.x * 256 + threadIdx.x;
  if (l >= 512) return;
  float acc = bsh[l];
  for (int l2 = 0; l2 < 512; ++l2) {
    acc += tba[l2] * Ws[(size_t)l2 * 512 + l];
    acc += tbv[l2] * Ws[(size_t)(512 + l2) * 512 + l];
  }
  btot[l] = acc;
}

__global__ void final_fc(const float* __restrict__ hT, const float* __restrict__ Cw,
                         const float* __restrict__ b_fc, float* __restrict__ out) {
  const int idx = threadIdx.x;           // 128 = 64 batches * 2 outputs
  if (idx >= 128) return;
  const int b = idx >> 1, j = idx & 1;
  float acc = b_fc[j];
  for (int i = 0; i < 512; ++i) acc += hT[(size_t)b * 512 + i] * Cw[(size_t)i * 2 + j];
  out[idx] = acc;
}

// ---------------------------------------------------------------------------
// Host launcher
// ---------------------------------------------------------------------------
static inline uint8_t* ws_take(uint8_t*& p, size_t bytes) {
  uint8_t* r = p;
  p += (bytes + 255) & ~(size_t)255;
  return r;
}

extern "C" void kernel_launch(void* const* d_in, const int* in_sizes, int n_in,
                              void* d_out, int out_size, void* d_ws, size_t ws_size,
                              hipStream_t stream) {
  (void)in_sizes; (void)n_in; (void)out_size; (void)ws_size;
  const float* audio    = (const float*)d_in[0];
  const float* video    = (const float*)d_in[1];
  const float* w_aconv  = (const float*)d_in[2];
  const float* b_aconv  = (const float*)d_in[3];
  const float* w_vconv  = (const float*)d_in[4];
  const float* b_vconv  = (const float*)d_in[5];
  const float* w_alat   = (const float*)d_in[6];
  const float* b_alat   = (const float*)d_in[7];
  const float* w_vlat   = (const float*)d_in[8];
  const float* b_vlat   = (const float*)d_in[9];
  const float* w_shared = (const float*)d_in[10];
  const float* b_shared = (const float*)d_in[11];
  const float* bn_gamma = (const float*)d_in[12];
  const float* bn_beta  = (const float*)d_in[13];
  const float* w_res    = (const float*)d_in[14];
  const float* b_res    = (const float*)d_in[15];
  const float* A_state  = (const float*)d_in[16];
  const float* B_in     = (const float*)d_in[17];
  const float* C_out    = (const float*)d_in[18];
  const float* w_fc     = (const float*)d_in[19];
  const float* b_fc     = (const float*)d_in[20];

  uint8_t* p = (uint8_t*)d_ws;
  float*  T1a  = (float*) ws_take(p, (size_t)128 * 512 * 4);
  float*  T1v  = (float*) ws_take(p, (size_t)512 * 512 * 4);
  float*  Ua   = (float*) ws_take(p, (size_t)128 * 512 * 4);
  float*  Uv   = (float*) ws_take(p, (size_t)512 * 512 * 4);
  float*  tba  = (float*) ws_take(p, 512 * 4);
  float*  tbv  = (float*) ws_take(p, 512 * 4);
  float*  btot = (float*) ws_take(p, 512 * 4);
  bf16_t* WtF  = (bf16_t*)ws_take(p, (size_t)512 * 640 * 2);
  bf16_t* WrT  = (bf16_t*)ws_take(p, (size_t)512 * 1536 * 2);
  bf16_t* BinT = (bf16_t*)ws_take(p, (size_t)512 * 512 * 2);
  bf16_t* MT   = (bf16_t*)ws_take(p, (size_t)512 * 512 * 2);
  float*  Mb0  = (float*) ws_take(p, (size_t)512 * 512 * 4);
  float*  Mb1  = (float*) ws_take(p, (size_t)512 * 512 * 4);
  float*  Cw   = (float*) ws_take(p, (size_t)512 * 2 * 4);
  bf16_t* Hbf  = (bf16_t*)ws_take(p, (size_t)65536 * 512 * 2);  // reused as V ping
  bf16_t* Xbf  = (bf16_t*)ws_take(p, (size_t)65536 * 512 * 2);
  float*  Ubuf = (float*) ws_take(p, (size_t)1024 * 64 * 512 * 4);
  float*  Vping = (float*)Hbf;  // 512*64*512*4 = 64 MiB fits exactly in Hbf

  // --- fold front linear layers into one K=640 weight ---
  gemm_f32<<<dim3((128 * 512 + 255) / 256), 256, 0, stream>>>(w_aconv, w_alat, T1a, 128, 512, 512, 1, 128, 512, 1);
  gemm_f32<<<dim3((512 * 512 + 255) / 256), 256, 0, stream>>>(w_vconv, w_vlat, T1v, 512, 512, 512, 1, 512, 512, 1);
  gemm_f32<<<dim3((128 * 512 + 255) / 256), 256, 0, stream>>>(T1a, w_shared, Ua, 128, 512, 512, 512, 1, 512, 1);
  gemm_f32<<<dim3((512 * 512 + 255) / 256), 256, 0, stream>>>(T1v, w_shared + (size_t)512 * 512, Uv, 512, 512, 512, 512, 1, 512, 1);
  bias_mm<<<dim3(2), 256, 0, stream>>>(b_aconv, w_alat, b_alat, tba, 512, 512);
  bias_mm<<<dim3(2), 256, 0, stream>>>(b_vconv, w_vlat, b_vlat, tbv, 512, 512);
  bias_combine<<<dim3(2), 256, 0, stream>>>(tba, tbv, w_shared, b_shared, btot);
  pack_frontT<<<dim3((512 * 640 + 255) / 256), 256, 0, stream>>>(Ua, Uv, WtF);
  pack_convT<<<dim3((512 * 1536 + 255) / 256), 256, 0, stream>>>(w_res, WrT);
  packT_bf16<<<dim3((512 * 512 + 255) / 256), 256, 0, stream>>>(B_in, BinT, 512, 512);
  gemm_f32<<<dim3((512 * 2 + 255) / 256), 256, 0, stream>>>(C_out, w_fc, Cw, 512, 2, 512, 512, 1, 2, 1);

  // --- big WMMA GEMMs ---
  dim3 gmain(65536 / BLK_M, 512 / BLK_N);
  front_gemm<<<gmain, 128, 0, stream>>>(audio, video, WtF, btot, bn_gamma, bn_beta, Hbf);
  conv_gemm<<<gmain, 128, 0, stream>>>(Hbf, WrT, b_res, Xbf);
  u_gemm<<<gmain, 128, 0, stream>>>(Xbf, BinT, Ubuf);

  // --- log-depth tree fold of the recurrence: h_T = sum_t U_t A^(T-1-t) ---
  const float* Mcur = A_state;
  float* Mbufs[2] = {Mb0, Mb1};
  int mb = 0;
  float* Vs = Ubuf;
  float* Vd = Vping;
  for (int lvl = 0; lvl < 10; ++lvl) {
    packT_bf16<<<dim3((512 * 512 + 255) / 256), 256, 0, stream>>>(Mcur, MT, 512, 512);
    const int nseg = 512 >> lvl;  // output segments at this level
    combine_gemm<<<dim3(nseg, 8), 128, 0, stream>>>(Vs, MT, Vd);
    { float* t = Vs; Vs = Vd; Vd = t; }
    if (lvl < 9) {
      float* Mn = Mbufs[mb]; mb ^= 1;
      gemm_f32<<<dim3((512 * 512 + 255) / 256), 256, 0, stream>>>(Mcur, Mcur, Mn, 512, 512, 512, 512, 1, 512, 1);
      Mcur = Mn;
    }
  }
  // after level 9 (even number of swaps ends on Ubuf), Vs[0..64*512) = h_T
  final_fc<<<dim3(1), 128, 0, stream>>>(Vs, Cw, b_fc, (float*)d_out);
}
